// BindingStrength_23270132810218
// MI455X (gfx1250) — matrix operand order repaired and checked by reading, not compiled, not run on hardware
//
#include <hip/hip_runtime.h>
#include <hip/hip_bf16.h>

typedef __attribute__((ext_vector_type(16))) _Float16 v16h;
typedef __attribute__((ext_vector_type(8)))  _Float16 v8h;
typedef __attribute__((ext_vector_type(8)))  float    v8f;
typedef __attribute__((ext_vector_type(4)))  int      v4i;

#define NA     8192
#define NB     8192
#define NBATCH 64
#define NTILE  64      // N columns per workgroup
#define KSTEP  64      // K per staged tile (2 WMMA k-halves)
#define LDSP   72      // padded LDS pitch (f16 elems) per transposed column

// ---------------- kernel 1: prep (trig of phases_a -> f16 A, zero accums) ----
__global__ void plv_prep(const float* __restrict__ phases_a,
                         _Float16* __restrict__ A16,
                         float* __restrict__ ws_acc,        // real[64] | imag[64]
                         unsigned* __restrict__ ws_np) {
  const int idx = blockIdx.x * blockDim.x + threadIdx.x;    // 0 .. 64*8192-1
  if (idx < 2 * NBATCH) ws_acc[idx] = 0.0f;
  if (idx == 0) *ws_np = 0u;
  const int b = idx >> 13;          // batch row
  const int i = idx & (NA - 1);     // oscillator index (K dim)
  const float p = phases_a[idx];
  float s, c;
  sincosf(p, &s, &c);
  A16[(size_t)b * NA + i]            = (_Float16)c;   // rows 0..63  : cos
  A16[(size_t)(b + NBATCH) * NA + i] = (_Float16)s;   // rows 64..127: sin
}

// ---------------- kernel 2: fused masked-GEMM + reduction --------------------
__global__ void __launch_bounds__(256)
plv_gemm(const _Float16* __restrict__ A16,
         const int* __restrict__ mask,
         const float* __restrict__ phases_b,
         float* __restrict__ ws_real,
         float* __restrict__ ws_imag,
         unsigned* __restrict__ ws_np) {
  // transposed f16 tile [n][k], double buffered: 2 * 64 * 72 * 2B = 18 KB
  __shared__ __align__(16) unsigned short ldsB[2][NTILE * LDSP];

  const int tid  = threadIdx.x;
  const int lane = tid & 31;
  const int wave = tid >> 5;
  const int n0   = blockIdx.x * NTILE;

  int npsum = 0;

  // Cooperative mask tile load: 64(K) x 64(N) int32 -> f16 bits, transposed.
  // Thread t owns K rows {2kp, 2kp+1}, N cols nq..nq+7; the two K-adjacent
  // f16 values pack into one dword -> ds_store_b32.
  auto load_tile = [&](int buf, int k0) {
    const int kp = tid >> 3;          // 0..31  (k rows 2kp, 2kp+1)
    const int nq = (tid & 7) * 8;     // 0..56
    const size_t base = (size_t)(k0 + 2 * kp) * NB + n0 + nq;
    // non-temporal: 268 MB stream-once operand, keep it out of L2's way
    const v4i r0a = __builtin_nontemporal_load((const v4i*)(mask + base));
    const v4i r0b = __builtin_nontemporal_load((const v4i*)(mask + base + 4));
    const v4i r1a = __builtin_nontemporal_load((const v4i*)(mask + base + NB));
    const v4i r1b = __builtin_nontemporal_load((const v4i*)(mask + base + NB + 4));
    const int e0[8] = {r0a[0], r0a[1], r0a[2], r0a[3], r0b[0], r0b[1], r0b[2], r0b[3]};
    const int e1[8] = {r1a[0], r1a[1], r1a[2], r1a[3], r1b[0], r1b[1], r1b[2], r1b[3]};
    unsigned* dst32 = (unsigned*)&ldsB[buf][0];
#pragma unroll
    for (int j = 0; j < 8; ++j) {
      npsum += e0[j] + e1[j];
      const unsigned pk = (unsigned)(e0[j] * 0x3C00) |
                          ((unsigned)(e1[j] * 0x3C00) << 16);
      // halves index (nq+j)*LDSP + 2kp  ->  dword index (nq+j)*(LDSP/2) + kp
      dst32[(nq + j) * (LDSP / 2) + kp] = pk;
    }
    // prefetch two tiles ahead (covers HBM latency beyond the double buffer)
    if (k0 + 2 * KSTEP < NA) {
      __builtin_prefetch(mask + base + (size_t)2 * KSTEP * NB, 0, 0);
      __builtin_prefetch(mask + base + (size_t)2 * KSTEP * NB + NB, 0, 0);
    }
  };

  // A fragment addressing (16x32 f16): lane&15 = row, lane>>4 selects K half (16 contiguous f16).
  const int r0 = wave * 16;  // this wave's output-row base (0..112)
  const _Float16* abase =
      A16 + (size_t)(r0 + (lane & 15)) * NA + ((lane >> 4) * 16);

  v8f acc0 = {}, acc1 = {}, acc2 = {}, acc3 = {};

  load_tile(0, 0);
  __syncthreads();

  for (int kt = 0; kt < NA / KSTEP; ++kt) {      // 128 iterations
    if (kt + 1 < NA / KSTEP) load_tile((kt + 1) & 1, (kt + 1) * KSTEP);

    const unsigned short* bb = &ldsB[kt & 1][0];
    const int k0 = kt * KSTEP;
#pragma unroll
    for (int ks = 0; ks < 2; ++ks) {             // two 32-K halves of the tile
      const v8h alo = *(const v8h*)(abase + k0 + ks * 32);
      const v8h ahi = *(const v8h*)(abase + k0 + ks * 32 + 8);
      const v16h a = __builtin_shufflevector(
          alo, ahi, 0, 1, 2, 3, 4, 5, 6, 7, 8, 9, 10, 11, 12, 13, 14, 15);

      // B fragment: lane&15 = column, lane>>4 = K half (16 contiguous f16).
      const int cbase = (lane & 15) * LDSP + ks * 32 + (lane >> 4) * 16;

      const v8h b0l = *(const v8h*)(bb + cbase + 0 * 16 * LDSP);
      const v8h b0h = *(const v8h*)(bb + cbase + 0 * 16 * LDSP + 8);
      const v8h b1l = *(const v8h*)(bb + cbase + 1 * 16 * LDSP);
      const v8h b1h = *(const v8h*)(bb + cbase + 1 * 16 * LDSP + 8);
      const v8h b2l = *(const v8h*)(bb + cbase + 2 * 16 * LDSP);
      const v8h b2h = *(const v8h*)(bb + cbase + 2 * 16 * LDSP + 8);
      const v8h b3l = *(const v8h*)(bb + cbase + 3 * 16 * LDSP);
      const v8h b3h = *(const v8h*)(bb + cbase + 3 * 16 * LDSP + 8);
      const v16h b0 = __builtin_shufflevector(b0l, b0h, 0,1,2,3,4,5,6,7,8,9,10,11,12,13,14,15);
      const v16h b1 = __builtin_shufflevector(b1l, b1h, 0,1,2,3,4,5,6,7,8,9,10,11,12,13,14,15);
      const v16h b2 = __builtin_shufflevector(b2l, b2h, 0,1,2,3,4,5,6,7,8,9,10,11,12,13,14,15);
      const v16h b3 = __builtin_shufflevector(b3l, b3h, 0,1,2,3,4,5,6,7,8,9,10,11,12,13,14,15);

      acc0 = __builtin_amdgcn_wmma_f32_16x16x32_f16(false, a, false, b0,
                                                    (short)0, acc0, false, false);
      acc1 = __builtin_amdgcn_wmma_f32_16x16x32_f16(false, a, false, b1,
                                                    (short)0, acc1, false, false);
      acc2 = __builtin_amdgcn_wmma_f32_16x16x32_f16(false, a, false, b2,
                                                    (short)0, acc2, false, false);
      acc3 = __builtin_amdgcn_wmma_f32_16x16x32_f16(false, a, false, b3,
                                                    (short)0, acc3, false, false);
    }
    __syncthreads();
  }

  // ---- n_pairs: wave reduce then one atomic per wave ----
#pragma unroll
  for (int m = 16; m >= 1; m >>= 1) npsum += __shfl_xor(npsum, m, 32);
  if (lane == 0) atomicAdd(ws_np, (unsigned)npsum);

  // ---- fold C tiles into real/imag with cos/sin(phases_b) on the fly ----
  // C layout: element g of acc -> row r0 + (lane>>4)*8 + g, col = lane&15.
  const bool isSin = (r0 >= NBATCH);
  const int mrow = (lane >> 4) * 8;
#pragma unroll
  for (int g = 0; g < 8; ++g) {
    const int b = r0 + mrow + g - (isSin ? NBATCH : 0);   // batch index
    const float Cv[4] = {acc0[g], acc1[g], acc2[g], acc3[g]};
    float re = 0.f, im = 0.f;
#pragma unroll
    for (int cb = 0; cb < 4; ++cb) {
      const int n = n0 + cb * 16 + (lane & 15);
      const float pb = phases_b[(size_t)b * NB + n];
      float sv, cv;
      sincosf(pb, &sv, &cv);
      const float C = Cv[cb];
      if (!isSin) { re += C * cv; im -= C * sv; }   // caM: +C*cb, -C*sb
      else        { re += C * sv; im += C * cv; }   // saM: +C*sb, +C*cb
    }
    // reduce over the 16 lanes of this half-wave (same b, different n)
#pragma unroll
    for (int m = 1; m <= 8; m <<= 1) {
      re += __shfl_xor(re, m, 32);
      im += __shfl_xor(im, m, 32);
    }
    if ((lane & 15) == 0) {
      atomicAdd(&ws_real[b], re);
      atomicAdd(&ws_imag[b], im);
    }
  }
}

// ---------------- kernel 3: finalize ----------------------------------------
__global__ void plv_final(const float* __restrict__ ws_real,
                          const float* __restrict__ ws_imag,
                          const unsigned* __restrict__ ws_np,
                          float* __restrict__ out) {
  const int b = threadIdx.x;
  if (b < NBATCH) {
    const float r = ws_real[b];
    const float i = ws_imag[b];
    const float np = fmaxf((float)(*ws_np), 1.0f);
    out[b] = sqrtf(r * r + i * i) / np;
  }
}

// ---------------- host launcher ---------------------------------------------
extern "C" void kernel_launch(void* const* d_in, const int* in_sizes, int n_in,
                              void* d_out, int out_size, void* d_ws, size_t ws_size,
                              hipStream_t stream) {
  const float* phases_a = (const float*)d_in[0];   // (64, 8192) f32
  const float* phases_b = (const float*)d_in[1];   // (64, 8192) f32
  const int*   mask     = (const int*)d_in[2];     // (8192, 8192) i32
  float* out = (float*)d_out;                      // (64,) f32

  // workspace layout: real[64] | imag[64] | np (u32) | pad to 1KB | A16 (2MB)
  float*    ws_real = (float*)d_ws;
  float*    ws_imag = ws_real + NBATCH;
  unsigned* ws_np   = (unsigned*)(ws_imag + NBATCH);
  _Float16* A16     = (_Float16*)((char*)d_ws + 1024);

  plv_prep<<<(NBATCH * NA) / 256, 256, 0, stream>>>(phases_a, A16, ws_real, ws_np);
  plv_gemm<<<NB / NTILE, 256, 0, stream>>>(A16, mask, phases_b,
                                           ws_real, ws_imag, ws_np);
  plv_final<<<1, 64, 0, stream>>>(ws_real, ws_imag, ws_np, out);
}
